// Joint_82721070121385
// MI455X (gfx1250) — matrix operand order repaired and checked by reading, not compiled, not run on hardware
//
#include <hip/hip_runtime.h>
#include <hip/hip_bf16.h>
#include <math.h>

// MI455X / gfx1250, wave32. RNN-T joint network:
//   z = pred_proj[n,u,:] + enc_proj[n,t,:] + b1 ; x = tanh(z)
//   logits = x @ W2 + b2 ; out = log_softmax(logits)  (TEMP = 1)
//
// Dominant GEMM (80000x1024x2048, 335 GFLOP) on v_wmma_f32_16x16x32_bf16.
// 32 rows per wave (two 16-row groups = two consecutive t values): each B
// fragment from L2 feeds TWO WMMAs from registers (~10 GB L2 traffic,
// 32 FLOP/byte). Accumulators use CDNA5 >256-VGPR wave32 (s_set_vgpr_msb).
// v3: hardware transcendentals — v_tanh_f32 for the 82M-element tanh stage,
// v_exp_f32/v_log_f32 (__expf/__logf) for the 164M-element softmax — the
// libm versions were ~2G extra VALU ops, comparable to the GEMM itself.

typedef __attribute__((ext_vector_type(16))) __bf16 v16bf;
typedef __attribute__((ext_vector_type(8)))  float  v8f;

#define NN 4
#define TT 250
#define UU 80
#define HH 512
#define JJ 1024
#define VV 2048

static __device__ __forceinline__ unsigned short f32_to_bf16_rne(float f) {
  unsigned int u = __float_as_uint(f);
  unsigned int r = u + 0x7fffu + ((u >> 16) & 1u);   // round-to-nearest-even
  return (unsigned short)(r >> 16);
}

static __device__ __forceinline__ float fast_tanh(float x) {
#if __has_builtin(__builtin_amdgcn_tanhf)
  return __builtin_amdgcn_tanhf(x);           // v_tanh_f32, hazards by compiler
#else
  float r;
  // v_nop satisfies the TRANS-op co-execution hazard (ISA 5.7.4) since the
  // compiler cannot see that the asm is a multicycle transcendental.
  asm("v_tanh_f32 %0, %1\n\tv_nop" : "=v"(r) : "v"(x));
  return r;
#endif
}

// ---------------------------------------------------------------------------
// Kernel 1: W2 [J,V] fp32 row-major  ->  W2b [V,J] bf16 (K contiguous).
// ---------------------------------------------------------------------------
__global__ __launch_bounds__(256) void convert_w2_kernel(
    const float* __restrict__ W2, unsigned short* __restrict__ W2b) {
  int e = blockIdx.x * 256 + threadIdx.x;   // e in [0, V*J)
  int v = e >> 10;
  int j = e & (JJ - 1);
  W2b[(size_t)v * JJ + j] = f32_to_bf16_rne(W2[(size_t)j * VV + v]);
}

// ---------------------------------------------------------------------------
// Kernel 2: small projections (fp32 FMA; 1.4 GFLOP total, negligible).
//   rows [0,320):    predp[row] = h_pred[row] @ W1[0:512]
//   rows [320,1320): encb[row-320] = h_enc[row-320] @ W1[512:1024] + b1
// ---------------------------------------------------------------------------
__global__ __launch_bounds__(256) void proj_kernel(
    const float* __restrict__ h_pred, const float* __restrict__ h_enc,
    const float* __restrict__ W1, const float* __restrict__ b1,
    float* __restrict__ predp, float* __restrict__ encb) {
  __shared__ float sh[HH];
  int row = blockIdx.x >> 2;
  int j   = ((blockIdx.x & 3) << 8) + threadIdx.x;
  const float* src;
  const float* w;
  float*       dst;
  float bias = 0.0f;
  if (row < NN * UU) {
    src = h_pred + (size_t)row * HH;
    w   = W1;
    dst = predp + (size_t)row * JJ;
  } else {
    int r = row - NN * UU;
    src = h_enc + (size_t)r * HH;
    w   = W1 + (size_t)HH * JJ;
    dst = encb + (size_t)r * JJ;
    bias = b1[j];
  }
  sh[threadIdx.x]       = src[threadIdx.x];
  sh[threadIdx.x + 256] = src[threadIdx.x + 256];
  __syncthreads();
  float acc = bias;
#pragma unroll 8
  for (int h = 0; h < HH; ++h)
    acc = fmaf(sh[h], w[(size_t)h * JJ + j], acc);
  dst[j] = acc;
}

// ---------------------------------------------------------------------------
// Kernel 3: fused tanh + GEMM (WMMA bf16) + log_softmax + store.
// Block = 256 threads (8 waves) owns 32 rows (t0,t0+1; 16 u's) x 2048 cols.
// Each wave: 32 rows x 256 cols -> 2x16 v8f accumulator tiles (256 VGPRs).
// A tiles (2 x 16x1024 bf16 = 64 KB LDS) pre-permuted to the WMMA A-fragment
// layout (contiguous k = 16h + s, as B): lanes 0-15 hold K 0-15 of row m, lanes 16-31
// hold K 16-31 -> one contiguous 32 B ds read per fragment.
// B fragment: lane l (<16) = col n0+l, K k0..k0+15; lane l+16 = K k0+16..+31
// -> one contiguous 32 B global load from transposed bf16 W2b; reused for
// both row groups.
// ---------------------------------------------------------------------------
__global__ __launch_bounds__(256) void joint_kernel(
    const float* __restrict__ predp, const float* __restrict__ encb,
    const unsigned short* __restrict__ W2b, const float* __restrict__ b2,
    float* __restrict__ out) {
  __shared__ unsigned short ldsA[32 * 2 * 32 * 16];  // [kb][g][lane][16] = 64 KB
  __shared__ float red[8 * 32];                      // [wave][g*16 + row]

  const int b   = blockIdx.x;             // 2500 blocks
  const int n   = b / (125 * 5);
  const int rem = b % (125 * 5);
  const int t0  = (rem / 5) * 2;          // two consecutive t per block
  const int u0  = (rem % 5) * 16;
  const int tid = threadIdx.x;

  const float* prow = predp + (size_t)(n * UU + u0) * JJ;  // 16 pred rows
  const float* erow = encb  + (size_t)(n * TT + t0) * JJ;  // 2 enc rows (+b1)

  // ---- stage A = tanh(pred + enc + b1) into permuted LDS layout ----
#pragma unroll 4
  for (int i = 0; i < 128; ++i) {
    int e = i * 256 + tid;              // 2*16*1024 elements / 256 threads
    int g = e >> 14;                    // row group (t0+g)
    int m = (e >> 10) & 15;             // u row within group
    int j = e & (JJ - 1);
    float x = fast_tanh(prow[(size_t)m * JJ + j] + erow[(size_t)g * JJ + j]);
    int kb   = j >> 5;
    int kk   = j & 31;
    int hi   = kk >> 4;
    int pos  = kk & 15;
    int lane = m | (hi << 4);
    ldsA[((((kb << 1) | g) << 5 | lane) << 4) | pos] = f32_to_bf16_rne(x);
  }
  __syncthreads();

  const int wave = tid >> 5;
  const int lane = tid & 31;
  const int nlo  = lane & 15;
  const int colbase = wave * 256 + nlo;   // this wave: cols [wave*256, +256)

  v8f acc[2][16];
#pragma unroll
  for (int g = 0; g < 2; ++g)
#pragma unroll
    for (int i = 0; i < 16; ++i) acc[g][i] = (v8f){0, 0, 0, 0, 0, 0, 0, 0};

  const unsigned short* Bp0 = W2b + (size_t)colbase * JJ + ((lane >> 4) << 4);

  for (int kb = 0; kb < 32; ++kb) {       // K = 1024 in steps of 32
    v16bf a0 = *(const v16bf*)&ldsA[(((kb << 1) | 0) << 5 | lane) << 4];
    v16bf a1 = *(const v16bf*)&ldsA[(((kb << 1) | 1) << 5 | lane) << 4];
    const unsigned short* bp = Bp0 + kb * 32;
#pragma unroll
    for (int tn = 0; tn < 16; ++tn) {
      v16bf bb = *(const v16bf*)(bp + (size_t)tn * 16 * JJ);
      acc[0][tn] = __builtin_amdgcn_wmma_f32_16x16x32_bf16(
          false, a0, false, bb, (short)0, acc[0][tn], false, false);
      acc[1][tn] = __builtin_amdgcn_wmma_f32_16x16x32_bf16(
          false, a1, false, bb, (short)0, acc[1][tn], false, false);
    }
  }

  // ---- + b2 ----
#pragma unroll
  for (int tn = 0; tn < 16; ++tn) {
    float bv = b2[wave * 256 + tn * 16 + nlo];
#pragma unroll
    for (int g = 0; g < 2; ++g)
#pragma unroll
      for (int r = 0; r < 8; ++r) acc[g][tn][r] += bv;
  }

  // D layout: VGPR r, lane l -> row (r + 8*(l>>4)), col (l&15) + 16*tn.
  const int rowhalf = (lane >> 4) << 3;

  // ---- row max: lane-local, shfl over 16-lane half, LDS across waves ----
  float rmax[2][8];
#pragma unroll
  for (int g = 0; g < 2; ++g) {
#pragma unroll
    for (int r = 0; r < 8; ++r) {
      float m = acc[g][0][r];
#pragma unroll
      for (int tn = 1; tn < 16; ++tn) m = fmaxf(m, acc[g][tn][r]);
#pragma unroll
      for (int off = 8; off; off >>= 1) m = fmaxf(m, __shfl_xor(m, off, 16));
      rmax[g][r] = m;
    }
    if (nlo == 0) {
#pragma unroll
      for (int r = 0; r < 8; ++r)
        red[wave * 32 + g * 16 + rowhalf + r] = rmax[g][r];
    }
  }
  __syncthreads();
  float gmax[2][8];
#pragma unroll
  for (int g = 0; g < 2; ++g)
#pragma unroll
    for (int r = 0; r < 8; ++r) {
      float m = red[g * 16 + rowhalf + r];
#pragma unroll
      for (int w = 1; w < 8; ++w)
        m = fmaxf(m, red[w * 32 + g * 16 + rowhalf + r]);
      gmax[g][r] = m;
    }
  __syncthreads();

  // ---- row sum of exp(acc - gmax), native v_exp_f32 ----
  float lsum[2][8];
#pragma unroll
  for (int g = 0; g < 2; ++g) {
#pragma unroll
    for (int r = 0; r < 8; ++r) {
      float s = 0.0f;
#pragma unroll
      for (int tn = 0; tn < 16; ++tn) s += __expf(acc[g][tn][r] - gmax[g][r]);
#pragma unroll
      for (int off = 8; off; off >>= 1) s += __shfl_xor(s, off, 16);
      lsum[g][r] = s;
    }
    if (nlo == 0) {
#pragma unroll
      for (int r = 0; r < 8; ++r)
        red[wave * 32 + g * 16 + rowhalf + r] = lsum[g][r];
    }
  }
  __syncthreads();
#pragma unroll
  for (int g = 0; g < 2; ++g)
#pragma unroll
    for (int r = 0; r < 8; ++r) {
      float s = 0.0f;
#pragma unroll
      for (int w = 0; w < 8; ++w) s += red[w * 32 + g * 16 + rowhalf + r];
      lsum[g][r] = __logf(s);
    }

  // ---- write out = acc - gmax - log(sum) ----
#pragma unroll
  for (int g = 0; g < 2; ++g) {
    const size_t orow = (size_t)(n * TT + t0 + g) * UU + u0;
#pragma unroll
    for (int r = 0; r < 8; ++r) {
      int m = rowhalf + r;
      float sub = gmax[g][r] + lsum[g][r];
      float* op = out + (orow + m) * (size_t)VV + wave * 256 + nlo;
#pragma unroll
      for (int tn = 0; tn < 16; ++tn) op[tn * 16] = acc[g][tn][r] - sub;
    }
  }
}

// ---------------------------------------------------------------------------
extern "C" void kernel_launch(void* const* d_in, const int* in_sizes, int n_in,
                              void* d_out, int out_size, void* d_ws, size_t ws_size,
                              hipStream_t stream) {
  const float* h_pred = (const float*)d_in[0];   // [N,U,H]
  const float* h_enc  = (const float*)d_in[1];   // [N,T,H]
  const float* W1     = (const float*)d_in[2];   // [2H,J]
  const float* b1     = (const float*)d_in[3];   // [J]
  const float* W2     = (const float*)d_in[4];   // [J,V]
  const float* b2     = (const float*)d_in[5];   // [V]
  float* out = (float*)d_out;                    // [N,T,U,V]

  // workspace: W2b (bf16, 4 MB) | predp (1.25 MB) | encb (3.9 MB) ~ 9.4 MB
  unsigned short* W2b = (unsigned short*)d_ws;
  float* predp = (float*)((char*)d_ws + (size_t)VV * JJ * sizeof(unsigned short));
  float* encb  = predp + (size_t)NN * UU * JJ;

  convert_w2_kernel<<<(VV * JJ) / 256, 256, 0, stream>>>(W2, W2b);
  proj_kernel<<<(NN * UU + NN * TT) * 4, 256, 0, stream>>>(h_pred, h_enc, W1, b1,
                                                           predp, encb);
  joint_kernel<<<NN * (TT / 2) * (UU / 16), 256, 0, stream>>>(predp, encb, W2b,
                                                              b2, out);
}
